// ALIGNN_47158740910619
// MI455X (gfx1250) — matrix operand order repaired and checked by reading, not compiled, not run on hardware
//
#include <hip/hip_runtime.h>

// ---------------------------------------------------------------------------
// Types for CDNA5 WMMA (wave32, 16x16x32 bf16 -> f32 accumulate)
// ---------------------------------------------------------------------------
typedef __bf16 bf16_t;
typedef bf16_t v16bf __attribute__((ext_vector_type(16)));
typedef float  v8f   __attribute__((ext_vector_type(8)));
typedef int    vec4i __attribute__((vector_size(16)));   // for async-to-LDS builtin

#define N_NODES   100000
#define N_EDGES   1600000
#define F_IN      92
#define E_IN      50
#define D         64
#define N_LAYERS  3
#define N_GRAPHS  512
#define ECHUNK    200000
#define BN_EPS    1e-5f

// fast silu: x * rcp(1 + exp(-x))  (v_exp_f32 + v_rcp_f32, no IEEE divide)
__device__ __forceinline__ float silu_f(float x) {
  return x * __builtin_amdgcn_rcpf(1.0f + __expf(-x));
}

__device__ __forceinline__ v8f wmma_bf16(v16bf a, v16bf b, v8f c) {
  // (neg_a, A, neg_b, B, c_mod, C, reuse_a, reuse_b)
  return __builtin_amdgcn_wmma_f32_16x16x32_bf16(false, a, false, b, (short)0, c, false, false);
}

// A fragment: 16x32 bf16 tile from row-major LDS. rows = lane&15,
// K contiguous per lane half (k = 16h + s); two 16B LDS loads.
__device__ __forceinline__ v16bf load_frag_a(const bf16_t* base, int row_stride, int kc) {
  int lane = threadIdx.x & 31;
  int m  = lane & 15;
  int kb = (lane < 16) ? 0 : 16;
  const bf16_t* p = base + m * row_stride + kc + kb;
  union { uint4 u[2]; v16bf v; } fa;
  fa.u[0] = *(const uint4*)(p);        // K = kc+kb .. kc+kb+7
  fa.u[1] = *(const uint4*)(p + 8);    // K = kc+kb+8 .. +7
  return fa.v;
}

// B fragment from pre-swizzled LDS: fragment fragIdx occupies
// ((fragIdx*32 + lane)*16) contiguous bf16 per lane -> two ds_load_b128.
__device__ __forceinline__ v16bf load_frag_sw(const bf16_t* base, int fragIdx) {
  int lane = threadIdx.x & 31;
  const bf16_t* p = base + (fragIdx * 32 + lane) * 16;
  union { uint4 u[2]; v16bf v; } fb;
  fb.u[0] = *(const uint4*)(p);
  fb.u[1] = *(const uint4*)(p + 8);
  return fb.v;
}

union Frag8 { uint4 u; bf16_t h[8]; };

// ---------------------------------------------------------------------------
// 0) zero helper
// ---------------------------------------------------------------------------
__global__ void k_zero(float* __restrict__ p, int n) {
  int i = blockIdx.x * 256 + threadIdx.x;
  if (i < n) p[i] = 0.0f;
}

// ---------------------------------------------------------------------------
// 1) pre-FC: h = silu(x @ pre_w + pre_b), output bf16 [N, 64]
//    K padded 92->96 (3 k-chunks), N=64 (4 n-tiles).
// ---------------------------------------------------------------------------
__global__ __launch_bounds__(256) void k_prefc(const float* __restrict__ x,
                                               const float* __restrict__ pre_w,
                                               const float* __restrict__ pre_b,
                                               bf16_t* __restrict__ hbf) {
  __shared__ __align__(16) bf16_t sA[128 * 96];
  __shared__ __align__(16) bf16_t sB[96 * 64];    // swizzled: 3 kc x 4 nt frags
  __shared__ float sBias[64];
  int tid  = threadIdx.x;
  int row0 = blockIdx.x * 128;

  // swizzled B staging: idx = ((kcIdx*4 + nt)*32 + lane)*16 + i
  for (int idx = tid; idx < 96 * 64; idx += 256) {
    int i     = idx & 15;
    int lane  = (idx >> 4) & 31;
    int nt    = (idx >> 9) & 3;
    int kcIdx = idx >> 11;
    int k = kcIdx * 32 + ((lane & 16) ? 16 : 0) + i;
    int n = nt * 16 + (lane & 15);
    sB[idx] = (bf16_t)((k < F_IN) ? pre_w[k * D + n] : 0.0f);
  }
  if (tid < 64) sBias[tid] = pre_b[tid];
  for (int idx = tid; idx < 128 * 96; idx += 256) {
    int r = idx / 96, c = idx - r * 96;
    int row = row0 + r;
    float v = (row < N_NODES && c < F_IN) ? x[(size_t)row * F_IN + c] : 0.0f;
    sA[idx] = (bf16_t)v;
  }
  __syncthreads();

  int wave = tid >> 5, lane = tid & 31;
  bool full = (row0 + 128 <= N_NODES);
  const bf16_t* aBase = sA + (wave * 16) * 96;
  v16bf a0 = load_frag_a(aBase, 96, 0);
  v16bf a1 = load_frag_a(aBase, 96, 32);
  v16bf a2 = load_frag_a(aBase, 96, 64);
#pragma unroll
  for (int nt = 0; nt < 4; ++nt) {
    v8f c = {};
    c = wmma_bf16(a0, load_frag_sw(sB, 0 * 4 + nt), c);
    c = wmma_bf16(a1, load_frag_sw(sB, 1 * 4 + nt), c);
    c = wmma_bf16(a2, load_frag_sw(sB, 2 * 4 + nt), c);
    int col  = nt * 16 + (lane & 15);
    float bias = sBias[col];
    int rbase = row0 + wave * 16 + ((lane < 16) ? 0 : 8);
    if (full) {
#pragma unroll
      for (int r = 0; r < 8; ++r)
        hbf[(rbase + r) * D + col] = (bf16_t)silu_f(c[r] + bias);
    } else {
#pragma unroll
      for (int r = 0; r < 8; ++r) {
        int row = rbase + r;
        if (row < N_NODES) hbf[row * D + col] = (bf16_t)silu_f(c[r] + bias);
      }
    }
  }
}

// ---------------------------------------------------------------------------
// 2) node GEMM: [k|q|v|skip] = h @ [Wk|Wq|Wv|Wskip] (+biases).
//    k,q,v -> bf16; skip+conv_bias -> fp32 acc (scatter accumulator init).
//    K=64 (2 k-chunks), N=256 (16 n-tiles).
//    A tile staged with CDNA5 async global->LDS loads when available.
// ---------------------------------------------------------------------------
__global__ __launch_bounds__(256) void k_nodegemm(const bf16_t* __restrict__ hbf,
                                                  const float* __restrict__ Wk,
                                                  const float* __restrict__ bk,
                                                  const float* __restrict__ Wq,
                                                  const float* __restrict__ bq,
                                                  const float* __restrict__ Wv,
                                                  const float* __restrict__ bv,
                                                  const float* __restrict__ Wskip,
                                                  const float* __restrict__ cbias,
                                                  int layer,
                                                  bf16_t* __restrict__ kbf,
                                                  bf16_t* __restrict__ qbf,
                                                  bf16_t* __restrict__ vbf,
                                                  float* __restrict__ acc) {
  __shared__ __align__(16) bf16_t sA[128 * 64];
  __shared__ __align__(16) bf16_t sB[64 * 256];   // swizzled: 2 kc x 16 nt frags
  __shared__ float sBias[256];
  int tid  = threadIdx.x;
  int row0 = blockIdx.x * 128;
  int wofs = layer * D * D;

  // --- A tile staging: 128 rows x 64 bf16 = 1024 x 16B ---------------------
#if __has_builtin(__builtin_amdgcn_global_load_async_to_lds_b128)
  for (int idx = tid; idx < 1024; idx += 256) {
    int r = idx >> 3;
    int row = row0 + r;
    if (row >= N_NODES) row = 0;   // tail rows clamped; their C rows are never stored
    __builtin_amdgcn_global_load_async_to_lds_b128(
        (vec4i*)(hbf + (size_t)row * D + (idx & 7) * 8),
        (vec4i*)(sA + (size_t)idx * 8),
        0, 0);
  }
#else
  for (int idx = tid; idx < 1024; idx += 256) {
    int r = idx >> 3;
    int row = row0 + r;
    uint4 val = make_uint4(0u, 0u, 0u, 0u);
    if (row < N_NODES) val = ((const uint4*)(hbf + row * D))[idx & 7];
    ((uint4*)sA)[idx] = val;
  }
#endif

  // swizzled B staging: idx = ((kcIdx*16 + nt)*32 + lane)*16 + i
  for (int idx = tid; idx < 64 * 256; idx += 256) {
    int i     = idx & 15;
    int lane  = (idx >> 4) & 31;
    int nt    = (idx >> 9) & 15;
    int kcIdx = idx >> 13;
    int k = kcIdx * 32 + ((lane & 16) ? 16 : 0) + i;
    int n = nt * 16 + (lane & 15);
    int sel = n >> 6, col = n & 63;
    const float* wp = (sel == 0) ? Wk : (sel == 1) ? Wq : (sel == 2) ? Wv : Wskip;
    sB[idx] = (bf16_t)wp[wofs + k * D + col];
  }
  {
    int sel = tid >> 6, col = tid & 63;
    const float* bp = (sel == 0) ? bk : (sel == 1) ? bq : (sel == 2) ? bv : cbias;
    sBias[tid] = bp[layer * D + col];
  }

#if __has_builtin(__builtin_amdgcn_global_load_async_to_lds_b128)
#if __has_builtin(__builtin_amdgcn_s_wait_asynccnt)
  __builtin_amdgcn_s_wait_asynccnt(0);
#else
  asm volatile("s_wait_asynccnt 0x0" ::: "memory");
#endif
#endif
  __syncthreads();

  int wave = tid >> 5, lane = tid & 31;
  bool full = (row0 + 128 <= N_NODES);
  const bf16_t* aBase = sA + (wave * 16) * D;
  v16bf a0 = load_frag_a(aBase, D, 0);
  v16bf a1 = load_frag_a(aBase, D, 32);
#pragma unroll
  for (int nt = 0; nt < 16; ++nt) {
    v8f c = {};
    c = wmma_bf16(a0, load_frag_sw(sB, nt), c);
    c = wmma_bf16(a1, load_frag_sw(sB, 16 + nt), c);
    int n = nt * 16 + (lane & 15);
    int sel = n >> 6, col = n & 63;     // sel is uniform per nt after unroll
    float bias = sBias[n];
    int rbase = row0 + wave * 16 + ((lane < 16) ? 0 : 8);
    if (full) {
#pragma unroll
      for (int r = 0; r < 8; ++r) {
        int row = rbase + r;
        float val = c[r] + bias;
        if      (sel == 0) kbf[row * D + col] = (bf16_t)val;
        else if (sel == 1) qbf[row * D + col] = (bf16_t)val;
        else if (sel == 2) vbf[row * D + col] = (bf16_t)val;
        else               acc[row * D + col] = val;
      }
    } else {
#pragma unroll
      for (int r = 0; r < 8; ++r) {
        int row = rbase + r;
        if (row >= N_NODES) continue;
        float val = c[r] + bias;
        if      (sel == 0) kbf[row * D + col] = (bf16_t)val;
        else if (sel == 1) qbf[row * D + col] = (bf16_t)val;
        else if (sel == 2) vbf[row * D + col] = (bf16_t)val;
        else               acc[row * D + col] = val;
      }
    }
  }
}

// ---------------------------------------------------------------------------
// 3) edge embedding GEMM (chunked): e = edge_attr @ We[l]  -> bf16 [chunk, 64]
//    K padded 50->64 (2 k-chunks), N=64 (4 n-tiles).
// ---------------------------------------------------------------------------
__global__ __launch_bounds__(256) void k_egemm(const float* __restrict__ edge_attr,
                                               const float* __restrict__ We,
                                               int layer, int chunk_base, int chunk_len,
                                               bf16_t* __restrict__ ebf) {
  __shared__ __align__(16) bf16_t sA[128 * 64];
  __shared__ __align__(16) bf16_t sB[64 * 64];    // swizzled: 2 kc x 4 nt frags
  int tid = threadIdx.x;
  int e0  = blockIdx.x * 128;                     // chunk-relative
  const float* W = We + (size_t)layer * E_IN * D;

  // swizzled B staging: idx = ((kcIdx*4 + nt)*32 + lane)*16 + i
  for (int idx = tid; idx < 64 * 64; idx += 256) {
    int i     = idx & 15;
    int lane  = (idx >> 4) & 31;
    int nt    = (idx >> 9) & 3;
    int kcIdx = idx >> 11;
    int k = kcIdx * 32 + ((lane & 16) ? 16 : 0) + i;
    int n = nt * 16 + (lane & 15);
    sB[idx] = (bf16_t)((k < E_IN) ? W[k * D + n] : 0.0f);
  }
  for (int idx = tid; idx < 128 * 64; idx += 256) {
    int r = idx >> 6, c = idx & 63;
    int el = e0 + r;
    float v = 0.0f;
    if (el < chunk_len && c < E_IN)
      v = edge_attr[(size_t)(chunk_base + el) * E_IN + c];
    sA[idx] = (bf16_t)v;
  }
  __syncthreads();

  int wave = tid >> 5, lane = tid & 31;
  bool full = (e0 + 128 <= chunk_len);
  const bf16_t* aBase = sA + (wave * 16) * D;
  v16bf a0 = load_frag_a(aBase, D, 0);
  v16bf a1 = load_frag_a(aBase, D, 32);
#pragma unroll
  for (int nt = 0; nt < 4; ++nt) {
    v8f c = {};
    c = wmma_bf16(a0, load_frag_sw(sB, nt), c);
    c = wmma_bf16(a1, load_frag_sw(sB, 4 + nt), c);
    int col = nt * 16 + (lane & 15);
    int rbase = e0 + wave * 16 + ((lane < 16) ? 0 : 8);
    if (full) {
#pragma unroll
      for (int r = 0; r < 8; ++r)
        ebf[(rbase + r) * D + col] = (bf16_t)c[r];
    } else {
#pragma unroll
      for (int r = 0; r < 8; ++r) {
        int el = rbase + r;
        if (el < chunk_len) ebf[el * D + col] = (bf16_t)c[r];
      }
    }
  }
}

// ---------------------------------------------------------------------------
// 4) gather-gate-scatter: 8 lanes/edge, 8 feats/lane, fp32 atomic scatter.
//    gate = silu(k[dst] + q[src] + 2e) ; msg = gate * (v[src] + e)
// ---------------------------------------------------------------------------
__global__ __launch_bounds__(256) void k_gather(const int* __restrict__ ei,
                                                const bf16_t* __restrict__ ebf,
                                                const bf16_t* __restrict__ kbf,
                                                const bf16_t* __restrict__ qbf,
                                                const bf16_t* __restrict__ vbf,
                                                float* __restrict__ acc,
                                                int chunk_base, int chunk_len) {
  int gt = blockIdx.x * 256 + threadIdx.x;
  int el = gt >> 3;
  if (el >= chunk_len) return;
  int f0 = (gt & 7) * 8;
  int eg = chunk_base + el;
  int src = ei[eg];
  int dst = ei[N_EDGES + eg];

  Frag8 e8, k8, q8, v8;
  e8.u = *(const uint4*)(ebf + (size_t)el  * D + f0);
  k8.u = *(const uint4*)(kbf + (size_t)dst * D + f0);
  q8.u = *(const uint4*)(qbf + (size_t)src * D + f0);
  v8.u = *(const uint4*)(vbf + (size_t)src * D + f0);

  float* ap = acc + (size_t)dst * D + f0;
#pragma unroll
  for (int i = 0; i < 8; ++i) {
    float ev  = (float)e8.h[i];
    float s   = (float)k8.h[i] + (float)q8.h[i] + 2.0f * ev;
    float msg = silu_f(s) * ((float)v8.h[i] + ev);
    atomicAdd(ap + i, msg);
  }
}

// ---------------------------------------------------------------------------
// 5) BatchNorm: per-feature sum / sumsq reduction, then normalize (+pool last)
// ---------------------------------------------------------------------------
__global__ __launch_bounds__(256) void k_bnreduce(const float* __restrict__ acc,
                                                  float* __restrict__ bnsum,
                                                  float* __restrict__ bnsq) {
  __shared__ float sS[256], sQ[256];
  int tid = threadIdx.x;
  int f = tid & 63;
  int rg = tid >> 6;               // 4 row groups per block
  float s = 0.0f, q = 0.0f;
  for (int row = blockIdx.x * 4 + rg; row < N_NODES; row += gridDim.x * 4) {
    float v = acc[(size_t)row * D + f];
    s += v; q += v * v;
  }
  sS[tid] = s; sQ[tid] = q;
  __syncthreads();
  if (tid < 64) {
    s = sS[tid] + sS[tid + 64] + sS[tid + 128] + sS[tid + 192];
    q = sQ[tid] + sQ[tid + 64] + sQ[tid + 128] + sQ[tid + 192];
    atomicAdd(&bnsum[f], s);
    atomicAdd(&bnsq[f], q);
  }
}

__global__ __launch_bounds__(256) void k_bnnorm(const float* __restrict__ acc,
                                                const float* __restrict__ bnsum,
                                                const float* __restrict__ bnsq,
                                                const float* __restrict__ gamma,
                                                const float* __restrict__ beta,
                                                int layer,
                                                bf16_t* __restrict__ hbf,
                                                int do_pool,
                                                const int* __restrict__ batch_idx,
                                                float* __restrict__ pool_sum,
                                                float* __restrict__ pool_cnt) {
  int gt = blockIdx.x * 256 + threadIdx.x;
  if (gt >= N_NODES * D) return;
  int row = gt >> 6, f = gt & 63;
  float mean = bnsum[f] * (1.0f / (float)N_NODES);
  float var  = bnsq[f]  * (1.0f / (float)N_NODES) - mean * mean;
  float inv  = rsqrtf(var + BN_EPS);
  float v = (acc[gt] - mean) * inv * gamma[layer * D + f] + beta[layer * D + f];
  hbf[gt] = (bf16_t)v;
  if (do_pool) {
    int g = batch_idx[row];
    atomicAdd(&pool_sum[g * D + f], v);
    if (f == 0) atomicAdd(&pool_cnt[g], 1.0f);
  }
}

// ---------------------------------------------------------------------------
// 6) pooled head: g = mean; out = silu(g@post_w + post_b) @ out_w + out_b
// ---------------------------------------------------------------------------
__global__ __launch_bounds__(64) void k_final(const float* __restrict__ pool_sum,
                                              const float* __restrict__ pool_cnt,
                                              const float* __restrict__ post_w,
                                              const float* __restrict__ post_b,
                                              const float* __restrict__ out_w,
                                              const float* __restrict__ out_b,
                                              float* __restrict__ out) {
  __shared__ float g[64];
  __shared__ float red[64];
  int gr = blockIdx.x, n = threadIdx.x;
  float cnt = fmaxf(pool_cnt[gr], 1.0f);
  g[n] = pool_sum[gr * D + n] / cnt;
  __syncthreads();
  float a = post_b[n];
#pragma unroll
  for (int k = 0; k < 64; ++k) a += g[k] * post_w[k * D + n];
  red[n] = silu_f(a) * out_w[n];
  __syncthreads();
  for (int s = 32; s > 0; s >>= 1) {
    if (n < s) red[n] += red[n + s];
    __syncthreads();
  }
  if (n == 0) out[gr] = red[0] + out_b[0];
}

// ---------------------------------------------------------------------------
// launcher
// ---------------------------------------------------------------------------
extern "C" void kernel_launch(void* const* d_in, const int* in_sizes, int n_in,
                              void* d_out, int out_size, void* d_ws, size_t ws_size,
                              hipStream_t stream) {
  (void)in_sizes; (void)n_in; (void)out_size; (void)ws_size;
  const float* x         = (const float*)d_in[0];
  const int*   ei        = (const int*)  d_in[1];
  const float* edge_attr = (const float*)d_in[2];
  const int*   batch_idx = (const int*)  d_in[3];
  const float* pre_w     = (const float*)d_in[4];
  const float* pre_b     = (const float*)d_in[5];
  const float* Wk        = (const float*)d_in[6];
  const float* bk        = (const float*)d_in[7];
  const float* Wq        = (const float*)d_in[8];
  const float* bq        = (const float*)d_in[9];
  const float* Wv        = (const float*)d_in[10];
  const float* bv        = (const float*)d_in[11];
  const float* We        = (const float*)d_in[12];
  const float* Wskip     = (const float*)d_in[13];
  const float* cbias     = (const float*)d_in[14];
  const float* gamma     = (const float*)d_in[15];
  const float* beta      = (const float*)d_in[16];
  const float* post_w    = (const float*)d_in[17];
  const float* post_b    = (const float*)d_in[18];
  const float* out_w     = (const float*)d_in[19];
  const float* out_b     = (const float*)d_in[20];
  float* out = (float*)d_out;

  // workspace layout (all offsets multiples of 256B)
  char* ws = (char*)d_ws;
  const size_t OFF_HBF  = 0;                         // bf16 [N,64]   12.8 MB
  const size_t OFF_KBF  = 12800000;                  // bf16 [N,64]
  const size_t OFF_QBF  = 25600000;                  // bf16 [N,64]
  const size_t OFF_VBF  = 38400000;                  // bf16 [N,64]
  const size_t OFF_ACC  = 51200000;                  // f32  [N,64]   25.6 MB
  const size_t OFF_EBF  = 76800000;                  // bf16 [ECHUNK,64] 25.6 MB
  const size_t OFF_BN   = 102400000;                 // f32 sum[64] + sumsq[64]
  const size_t OFF_POOL = OFF_BN + 512;              // f32 [512,64] then cnt[512]

  bf16_t* hbf = (bf16_t*)(ws + OFF_HBF);
  bf16_t* kbf = (bf16_t*)(ws + OFF_KBF);
  bf16_t* qbf = (bf16_t*)(ws + OFF_QBF);
  bf16_t* vbf = (bf16_t*)(ws + OFF_VBF);
  float*  acc = (float*) (ws + OFF_ACC);
  bf16_t* ebf = (bf16_t*)(ws + OFF_EBF);
  float*  bnsum    = (float*)(ws + OFF_BN);
  float*  bnsq     = bnsum + 64;
  float*  pool_sum = (float*)(ws + OFF_POOL);
  float*  pool_cnt = pool_sum + N_GRAPHS * D;

  const int NODE_BLOCKS = (N_NODES + 127) / 128;   // 782

  // zero pooling accumulators (d_ws is poisoned, never re-zeroed by harness)
  k_zero<<<(N_GRAPHS * D + N_GRAPHS + 255) / 256, 256, 0, stream>>>(
      pool_sum, N_GRAPHS * D + N_GRAPHS);

  k_prefc<<<NODE_BLOCKS, 256, 0, stream>>>(x, pre_w, pre_b, hbf);

  for (int l = 0; l < N_LAYERS; ++l) {
    k_nodegemm<<<NODE_BLOCKS, 256, 0, stream>>>(hbf, Wk, bk, Wq, bq, Wv, bv,
                                                Wskip, cbias, l, kbf, qbf, vbf, acc);
    for (int cb = 0; cb < N_EDGES; cb += ECHUNK) {
      int clen = (N_EDGES - cb < ECHUNK) ? (N_EDGES - cb) : ECHUNK;
      k_egemm<<<(clen + 127) / 128, 256, 0, stream>>>(edge_attr, We, l, cb, clen, ebf);
      k_gather<<<(clen * 8 + 255) / 256, 256, 0, stream>>>(ei, ebf, kbf, qbf, vbf,
                                                           acc, cb, clen);
    }
    k_zero<<<1, 256, 0, stream>>>(bnsum, 128);
    k_bnreduce<<<512, 256, 0, stream>>>(acc, bnsum, bnsq);
    int do_pool = (l == N_LAYERS - 1) ? 1 : 0;
    k_bnnorm<<<(N_NODES * D + 255) / 256, 256, 0, stream>>>(
        acc, bnsum, bnsq, gamma, beta, l, hbf, do_pool, batch_idx,
        pool_sum, pool_cnt);
  }

  k_final<<<N_GRAPHS, 64, 0, stream>>>(pool_sum, pool_cnt, post_w, post_b,
                                       out_w, out_b, out);
}
